// GCNNet_12945031430852
// MI455X (gfx1250) — matrix operand order repaired and checked by reading, not compiled, not run on hardware
//
#include <hip/hip_runtime.h>
#include <math.h>

typedef __attribute__((ext_vector_type(16))) _Float16 v16h;
typedef __attribute__((ext_vector_type(8)))  _Float16 v8h;
typedef __attribute__((ext_vector_type(8)))  float    v8f;

// ---------------------------------------------------------------------------
// gcn_norm: deg (with self loop) -> dis = rsqrt(deg)
// ---------------------------------------------------------------------------
__global__ void k_deg_init(unsigned* deg, int N) {
  int i = blockIdx.x * blockDim.x + threadIdx.x;
  if (i < N) deg[i] = 1u;  // self-loop contributes 1
}

__global__ void k_deg_count(const int* __restrict__ dst, unsigned* deg, int E) {
  int e = blockIdx.x * blockDim.x + threadIdx.x;
  if (e < E) atomicAdd(&deg[dst[e]], 1u);
}

__global__ void k_deg_finalize(const unsigned* __restrict__ deg, float* __restrict__ dis, int N) {
  int i = blockIdx.x * blockDim.x + threadIdx.x;
  if (i < N) dis[i] = rsqrtf((float)deg[i]);
}

// ---------------------------------------------------------------------------
// f32 -> f16 conversion (activations)
// ---------------------------------------------------------------------------
__global__ void k_f32_to_f16(const float* __restrict__ in, _Float16* __restrict__ out, int n) {
  int i = blockIdx.x * blockDim.x + threadIdx.x;
  if (i < n) out[i] = (_Float16)in[i];
}

// Weight convert + transpose: W[K x Nout] f32 -> Bt[Nout x K] f16
__global__ void k_w_transpose_f16(const float* __restrict__ W, _Float16* __restrict__ Bt,
                                  int K, int Nout) {
  int idx = blockIdx.x * blockDim.x + threadIdx.x;
  if (idx < K * Nout) {
    int k = idx / Nout;
    int n = idx - k * Nout;
    Bt[n * K + k] = (_Float16)W[idx];
  }
}

// ---------------------------------------------------------------------------
// WMMA GEMM: C[M x NOUT] = A[M x K] * B[K x NOUT], f16 in, f32 accumulate.
// B supplied transposed (Bt[NOUT x K]) so every lane's K-walk is contiguous
// (pure global_load_b128 clauses, no scalar packing).
// One wave computes a 16x64 strip of C: 4 accumulators, A reused 4x per
// k-step. K and NOUT are template params so the K-loop fully unrolls and all
// store offsets become compile-time immediates. Host guarantees M % 16 == 0,
// so no per-row bounds checks are needed.
// Lane layouts as loaded here (wave32, same contiguous k order for A and B):
//   A (16x32 f16): lane m=lane&15; lanes 0-15: elems = K 0..15;
//                  lanes 16-31: elems = K 16..31.
//   B (32x16 f16): lane n=lane&15; lanes 0-15: elems = K 0..15;
//                  lanes 16-31: elems = K 16..31.
//   D (16x16 f32): elem r -> row r + (lane>=16 ? 8 : 0), col = lane&15.
// ---------------------------------------------------------------------------
template <int K, int NOUT>
__global__ void k_gemm_wmma_f16(const _Float16* __restrict__ A,
                                const _Float16* __restrict__ Bt,
                                float* __restrict__ C, int M) {
  constexpr int NT = 4;               // 4 16-col tiles per wave -> 16x64 strip
  constexpr int GROUPS = NOUT / 64;   // 64-col groups per row of tiles
  int wave = (blockIdx.x * blockDim.x + threadIdx.x) >> 5;
  int lane = threadIdx.x & 31;
  int mt = wave / GROUPS;
  int gt = wave - mt * GROUPS;
  if (mt * 16 >= M) return;

  int m  = lane & 15;
  int hi = lane >> 4;

  const _Float16* Ap = A + (mt * 16 + m) * K + hi * 16;
  const _Float16* Bp = Bt + (gt * 64 + m) * K + hi * 16;

  v8f acc[NT] = {};

#pragma unroll
  for (int kk = 0; kk < K; kk += 32) {
    v8h a0 = *(const v8h*)(Ap + kk);       // K = kk + hi*16 + [0..7]
    v8h a1 = *(const v8h*)(Ap + kk + 8);   // K = kk + hi*16 + 8 + [0..7]
    v16h a = __builtin_shufflevector(a0, a1, 0, 1, 2, 3, 4, 5, 6, 7,
                                     8, 9, 10, 11, 12, 13, 14, 15);
#pragma unroll
    for (int t = 0; t < NT; ++t) {
      v16h b = *(const v16h*)(Bp + t * 16 * K + kk);  // K = kk + hi*16 + [0..15]
      acc[t] = __builtin_amdgcn_wmma_f32_16x16x32_f16(
          false, a, false, b, (short)0, acc[t], false, false);
    }
  }

  // Store: base pointer per tile, row offsets are compile-time (r * NOUT).
  int rbase = mt * 16 + hi * 8;
#pragma unroll
  for (int t = 0; t < NT; ++t) {
    float* Cp = C + rbase * NOUT + gt * 64 + t * 16 + m;
#pragma unroll
    for (int r = 0; r < 8; ++r) {
      Cp[r * NOUT] = acc[t][r];
    }
  }
}

// ---------------------------------------------------------------------------
// Aggregation: out = self-term, then scatter-add over edges.
// ---------------------------------------------------------------------------
__global__ void k_self(const float* __restrict__ h, const float* __restrict__ dis,
                       float* __restrict__ out, int n, int dshift) {
  int idx = blockIdx.x * blockDim.x + threadIdx.x;
  if (idx < n) {
    float s = dis[idx >> dshift];
    out[idx] = h[idx] * s * s;  // self-loop: norm = dis[i]^2
  }
}

__global__ void k_scatter(const int* __restrict__ src, const int* __restrict__ dst,
                          const float* __restrict__ dis,
                          const float* __restrict__ h, float* __restrict__ out,
                          int E, int d) {
  int wid  = (blockIdx.x * blockDim.x + threadIdx.x) >> 5;
  int lane = threadIdx.x & 31;
  if (wid >= E) return;
  int s = src[wid];
  int t = dst[wid];
  float w = dis[s] * dis[t];
  const float* hp = h + s * d;
  float* op = out + t * d;
#pragma unroll 4
  for (int c = lane; c < d; c += 32) {
    unsafeAtomicAdd(&op[c], hp[c] * w);  // native global_atomic_add_f32 (hits L2)
  }
}

// ---------------------------------------------------------------------------
// Bias + exact GELU -> f16 activations for next layer; final: bias -> f32 out.
// ---------------------------------------------------------------------------
__global__ void k_bias_gelu_f16(const float* __restrict__ in, const float* __restrict__ b,
                                _Float16* __restrict__ out, int n, int dmask) {
  int idx = blockIdx.x * blockDim.x + threadIdx.x;
  if (idx < n) {
    float x = in[idx] + b[idx & dmask];
    float g = 0.5f * x * (1.0f + erff(x * 0.70710678118654752f));
    out[idx] = (_Float16)g;
  }
}

__global__ void k_bias_out(const float* __restrict__ in, const float* __restrict__ b,
                           float* __restrict__ out, int n, int dmask) {
  int idx = blockIdx.x * blockDim.x + threadIdx.x;
  if (idx < n) out[idx] = in[idx] + b[idx & dmask];
}

// ---------------------------------------------------------------------------
extern "C" void kernel_launch(void* const* d_in, const int* in_sizes, int n_in,
                              void* d_out, int out_size, void* d_ws, size_t ws_size,
                              hipStream_t stream) {
  (void)n_in; (void)out_size; (void)ws_size;

  const float* x  = (const float*)d_in[0];
  const int*   ei = (const int*)d_in[1];
  const float* W1 = (const float*)d_in[2];
  const float* b1 = (const float*)d_in[3];
  const float* W2 = (const float*)d_in[4];
  const float* b2 = (const float*)d_in[5];
  const float* W3 = (const float*)d_in[6];
  const float* b3 = (const float*)d_in[7];

  const int D_IN = 64, D_H = 128, D_OUT = 64;
  const int N = in_sizes[0] / D_IN;  // N = 100000, multiple of 16
  const int E = in_sizes[1] / 2;
  const int* srcp = ei;      // edge_index[0]
  const int* dstp = ei + E;  // edge_index[1]

  // Workspace carve-up (256B aligned)
  char* p = (char*)d_ws;
  auto carve = [&](size_t bytes) -> char* {
    char* r = p;
    p += (bytes + 255) & ~(size_t)255;
    return r;
  };
  unsigned*  deg   = (unsigned*)carve((size_t)N * 4);
  float*     dis   = (float*)carve((size_t)N * 4);
  _Float16*  wf16  = (_Float16*)carve((size_t)D_H * D_H * 2);  // transposed W, max 128x128
  _Float16*  hf16  = (_Float16*)carve((size_t)N * D_H * 2);    // f16 activations
  float*     gemmO = (float*)carve((size_t)N * D_H * 4);       // GEMM output
  float*     aggO  = (float*)carve((size_t)N * D_H * 4);       // aggregated

  dim3 blk(256);
  auto grid1 = [](long n) { return dim3((unsigned)((n + 255) / 256)); };

  // gcn_norm
  k_deg_init<<<grid1(N), blk, 0, stream>>>(deg, N);
  k_deg_count<<<grid1(E), blk, 0, stream>>>(dstp, deg, E);
  k_deg_finalize<<<grid1(N), blk, 0, stream>>>(deg, dis, N);

  // x -> f16 activations
  k_f32_to_f16<<<grid1((long)N * D_IN), blk, 0, stream>>>(x, hf16, N * D_IN);

  auto layer = [&](int din, int dout, const float* W, const float* b, bool last) {
    k_w_transpose_f16<<<grid1((long)din * dout), blk, 0, stream>>>(W, wf16, din, dout);

    long mtiles = N / 16;                // M % 16 == 0 (N = 100000)
    long waves  = mtiles * (dout / 64);  // each wave does a 16x64 strip
    dim3 g = grid1(waves * 32);
    if (din == 64 && dout == 128)
      k_gemm_wmma_f16<64, 128><<<g, blk, 0, stream>>>(hf16, wf16, gemmO, N);
    else if (din == 128 && dout == 128)
      k_gemm_wmma_f16<128, 128><<<g, blk, 0, stream>>>(hf16, wf16, gemmO, N);
    else
      k_gemm_wmma_f16<128, 64><<<g, blk, 0, stream>>>(hf16, wf16, gemmO, N);

    int dshift = (dout == 128) ? 7 : 6;
    k_self<<<grid1((long)N * dout), blk, 0, stream>>>(gemmO, dis, aggO, N * dout, dshift);
    k_scatter<<<grid1((long)E * 32), blk, 0, stream>>>(srcp, dstp, dis, gemmO, aggO, E, dout);

    if (!last)
      k_bias_gelu_f16<<<grid1((long)N * dout), blk, 0, stream>>>(aggO, b, hf16, N * dout, dout - 1);
    else
      k_bias_out<<<grid1((long)N * dout), blk, 0, stream>>>(aggO, b, (float*)d_out, N * dout, dout - 1);
  };

  layer(D_IN, D_H, W1, b1, false);
  layer(D_H,  D_H, W2, b2, false);
  layer(D_H,  D_OUT, W3, b3, true);
}
